// DigitConvolutionalModel_37495064494201
// MI455X (gfx1250) — compile-verified, matrix operand order checked by reading, not run on hardware
//
#include <hip/hip_runtime.h>

typedef __bf16 v16bf __attribute__((ext_vector_type(16)));
typedef float  v8f   __attribute__((ext_vector_type(8)));

#define K1    676          // 26*26 conv outputs
#define KP    704          // K1 padded to 22 * 32
#define KT1   (KP / 32)    // 22 k-tiles for GEMM1
#define N1    128
#define N2P   16           // 10 padded to 16
#define ROWS  16           // samples per block

// ---------------- weight packing ----------------
// w1 [676,128] fp32 -> w1p [128][704] bf16, transposed (lane reads contiguous K), zero-padded K
__global__ __launch_bounds__(256) void pack_w1_kernel(const float* __restrict__ w1,
                                                      __bf16* __restrict__ w1p) {
    int i = blockIdx.x * 256 + threadIdx.x;       // over 128 * KP
    if (i >= N1 * KP) return;
    int n = i / KP;
    int k = i - n * KP;
    float v = (k < K1) ? w1[k * N1 + n] : 0.0f;
    w1p[i] = (__bf16)v;
}

// w2 [128,10] fp32 -> w2p [16][128] bf16, transposed, zero-padded N
__global__ __launch_bounds__(256) void pack_w2_kernel(const float* __restrict__ w2,
                                                      __bf16* __restrict__ w2p) {
    int i = blockIdx.x * 256 + threadIdx.x;       // over 16 * 128
    if (i >= N2P * N1) return;
    int n = i / N1;
    int k = i - n * N1;
    float v = (n < 10) ? w2[k * 10 + n] : 0.0f;
    w2p[i] = (__bf16)v;
}

// ---------------- fused conv + MLP ----------------
// block = 256 threads (8 waves), handles 16 samples; wave w owns h columns [16w, 16w+16)
__global__ __launch_bounds__(256) void fused_conv_mlp_kernel(
    const float* __restrict__ x,      // [B,784]
    const float* __restrict__ cw,     // [3,3]
    const float* __restrict__ b1,     // [128]
    const float* __restrict__ b2,     // [10]
    const __bf16* __restrict__ w1p,   // [128][704] bf16
    const __bf16* __restrict__ w2p,   // [16][128] bf16
    float* __restrict__ out)          // [B,10]
{
    __shared__ __bf16 apanel[ROWS * KP];     // conv activations, bf16, K-padded
    __shared__ __bf16 hpanel[ROWS * N1];     // hidden layer, bf16

    const int tid  = threadIdx.x;
    const int lane = tid & 31;
    const int wave = tid >> 5;
    const long row0 = (long)blockIdx.x * ROWS;

    // prefetch this block's slab of x into the cache hierarchy
    __builtin_prefetch(x + (row0 + (tid & 15)) * 784 + (tid >> 4) * 49, 0, 3);

    // conv weights (uniform across block -> scalar loads)
    const float w00 = cw[0], w01 = cw[1], w02 = cw[2];
    const float w10 = cw[3], w11 = cw[4], w12 = cw[5];
    const float w20 = cw[6], w21 = cw[7], w22 = cw[8];

    // zero-fill K padding of the A panel (cols 676..703)
    for (int i = tid; i < ROWS * (KP - K1); i += 256) {
        int r = i / (KP - K1);
        int c = i - r * (KP - K1);
        apanel[r * KP + K1 + c] = (__bf16)0.0f;
    }

    // 3x3 valid conv, 16 rows x 676 outputs, straight from global (heavy cache reuse)
    for (int i = tid; i < ROWS * K1; i += 256) {
        int r  = i / K1;
        int p  = i - r * K1;
        int oy = p / 26;
        int ox = p - oy * 26;
        const float* xr = x + (row0 + r) * 784 + oy * 28 + ox;
        float a = xr[0]  * w00 + xr[1]  * w01 + xr[2]  * w02
                + xr[28] * w10 + xr[29] * w11 + xr[30] * w12
                + xr[56] * w20 + xr[57] * w21 + xr[58] * w22;
        apanel[r * KP + p] = (__bf16)a;
    }
    __syncthreads();

    // ---- GEMM1: [16 x 704] @ [704 x 16] per wave, bf16 WMMA, f32 accum ----
    const int nr    = lane & 15;          // A row / B,C column within tile
    const int khalf = (lane >> 4) * 16;   // K half selected by upper lane group
    const int mhi   = (lane >> 4) * 8;    // C/D row offset for upper lane group

    v8f acc = {};
    const __bf16* wrow = w1p + (wave * 16 + nr) * KP + khalf;
#pragma unroll
    for (int kt = 0; kt < KT1; ++kt) {
        v16bf a = *(const v16bf*)&apanel[nr * KP + kt * 32 + khalf];
        v16bf b = *(const v16bf*)&wrow[kt * 32];
        acc = __builtin_amdgcn_wmma_f32_16x16x32_bf16(
                  false, a, false, b, (short)0, acc, false, false);
    }

    // bias + ReLU, requantize to bf16 h-panel
    const float bias1 = b1[wave * 16 + nr];
#pragma unroll
    for (int r = 0; r < 8; ++r) {
        float v = acc[r] + bias1;
        v = v > 0.0f ? v : 0.0f;
        hpanel[(r + mhi) * N1 + wave * 16 + nr] = (__bf16)v;
    }
    __syncthreads();

    // ---- GEMM2: [16 x 128] @ [128 x 16] on wave 0 only (wave-uniform branch) ----
    if (wave == 0) {
        v8f acc2 = {};
#pragma unroll
        for (int kt = 0; kt < 4; ++kt) {
            v16bf a = *(const v16bf*)&hpanel[nr * N1 + kt * 32 + khalf];
            v16bf b = *(const v16bf*)&w2p[nr * N1 + kt * 32 + khalf];
            acc2 = __builtin_amdgcn_wmma_f32_16x16x32_bf16(
                       false, a, false, b, (short)0, acc2, false, false);
        }
        if (nr < 10) {
            const float bias2 = b2[nr];
#pragma unroll
            for (int r = 0; r < 8; ++r) {
                out[(row0 + r + mhi) * 10 + nr] = acc2[r] + bias2;
            }
        }
    }
}

extern "C" void kernel_launch(void* const* d_in, const int* in_sizes, int n_in,
                              void* d_out, int out_size, void* d_ws, size_t ws_size,
                              hipStream_t stream) {
    const float* x  = (const float*)d_in[0];   // [B,784]
    const float* cw = (const float*)d_in[1];   // [3,3]
    const float* w1 = (const float*)d_in[2];   // [676,128]
    const float* b1 = (const float*)d_in[3];   // [128]
    const float* w2 = (const float*)d_in[4];   // [128,10]
    const float* b2 = (const float*)d_in[5];   // [10]
    float* out = (float*)d_out;

    const int B = in_sizes[0] / 784;           // 65536

    // workspace: packed bf16 weights
    __bf16* w1p = (__bf16*)d_ws;                         // 128*704*2 = 180,224 B
    __bf16* w2p = (__bf16*)((char*)d_ws + N1 * KP * 2);  // 16*128*2  =   4,096 B

    pack_w1_kernel<<<(N1 * KP + 255) / 256, 256, 0, stream>>>(w1, w1p);
    pack_w2_kernel<<<(N2P * N1 + 255) / 256, 256, 0, stream>>>(w2, w2p);

    fused_conv_mlp_kernel<<<B / ROWS, 256, 0, stream>>>(x, cw, b1, b2, w1p, w2p, out);
}
